// MLP_26156350833317
// MI455X (gfx1250) — matrix operand order repaired and checked by reading, not compiled, not run on hardware
//
#include <hip/hip_runtime.h>

typedef __attribute__((ext_vector_type(16))) _Float16 v16h;
typedef __attribute__((ext_vector_type(8)))  _Float16 v8h;
typedef __attribute__((ext_vector_type(8)))  float    v8f;
typedef __attribute__((address_space(3))) char lds_char_t;

// Problem constants (fixed by the reference)
constexpr int Bc = 4, Cc = 64, Hc = 128, Wc = 128, Qc = 65536;
constexpr int QT = 32;          // queries per workgroup
constexpr int HS_STRIDE = 264;  // 256 + 8 halves pad (bank-conflict free)
constexpr int WS_STRIDE = 40;   // 32 + 8 halves pad

// f16 transposed-weight bank layout in workspace (element offsets, halves)
constexpr size_t OFF_W0 = 0;                      // [256][96]
constexpr size_t OFF_W1 = OFF_W0 + 256 * 96;      // [256][256]
constexpr size_t OFF_W2 = OFF_W1 + 256 * 256;
constexpr size_t OFF_W3 = OFF_W2 + 256 * 256;
constexpr size_t OFF_W4 = OFF_W3 + 256 * 256;     // [16][256]

// Shared memory carve (bytes)
constexpr int SM_HS   = 0;                        // _Float16 [128][264] = 67584
constexpr int SM_WS0  = 67584;                    // _Float16 [256][40]  = 20480
constexpr int SM_WS1  = 88064;                    // _Float16 [256][40]  = 20480
constexpr int SM_BIAS = 108544;                   // float [4][256]      = 4096
constexpr int SM_B4   = 112640;                   // float [16]          = 64
constexpr int SM_AREA = 112704;                   // float [4][32]       = 512
constexpr int SM_PRED = 113216;                   // float [128][4]      = 2048
constexpr int SMEM_BYTES = 115264;

// ---------------- weight prep: fp32 [K][N] -> f16 transposed [Nrows][Kpad] -------------
__global__ void liif_prep_w(const float* __restrict__ w, _Float16* __restrict__ dst,
                            int K, int N, int Kpad, int Nrows) {
  int idx = blockIdx.x * blockDim.x + threadIdx.x;
  int total = Nrows * Kpad;
  if (idx >= total) return;
  int n = idx / Kpad, k = idx - n * Kpad;
  float v = (k < K && n < N) ? w[k * N + n] : 0.0f;
  dst[idx] = (_Float16)v;
}

// Async DMA of one W chunk row (64 B) per thread: global -> LDS, no VGPR staging.
// Tracked by ASYNCcnt; completion enforced with s_wait_asynccnt + barrier.
__device__ __forceinline__ void issue_wchunk_async(const _Float16* __restrict__ Wtg,
                                                   _Float16* wsBuf, int Kpad, int kc,
                                                   int t, int nrows) {
  if (t < nrows) {
    unsigned ldsoff =
        (unsigned)(unsigned long long)(lds_char_t*)(void*)&wsBuf[t * WS_STRIDE];
    unsigned goff = (unsigned)((t * Kpad + kc * 32) * 2);  // byte offset per lane
    unsigned long long gbase = (unsigned long long)Wtg;
    asm volatile(
        "global_load_async_to_lds_b128 %0, %1, %2\n\t"
        "global_load_async_to_lds_b128 %0, %1, %2 offset:16\n\t"
        "global_load_async_to_lds_b128 %0, %1, %2 offset:32\n\t"
        "global_load_async_to_lds_b128 %0, %1, %2 offset:48"
        :
        : "v"(ldsoff), "v"(goff), "s"(gbase)
        : "memory");
  }
}

__device__ __forceinline__ void wait_async_all() {
  asm volatile("s_wait_asynccnt 0" ::: "memory");
}

// ---------------- one MLP layer: Hs[128][K] @ W[K][Ncols] (+bias, relu) ----------------
template <int NKC, int NTILES, int NROWS, bool RELU>
__device__ __forceinline__ void mlp_layer(_Float16* Hs, _Float16* Ws0, _Float16* Ws1,
                                          const _Float16* __restrict__ Wtg,
                                          const float* biasRow, float* predOut,
                                          int wave, int lm, int lg, int t) {
  v8f acc[NTILES];
#pragma unroll
  for (int i = 0; i < NTILES; ++i)
    acc[i] = v8f{0.f, 0.f, 0.f, 0.f, 0.f, 0.f, 0.f, 0.f};

  // prologue: start streaming chunk 0 (ASYNCcnt is zero at layer entry)
  issue_wchunk_async(Wtg, Ws0, NKC * 32, 0, t, NROWS);

#pragma unroll
  for (int kc = 0; kc < NKC; ++kc) {
    wait_async_all();   // chunk kc landed in LDS (this wave's portion)
    __syncthreads();    // ... and everyone else's portion too
    if (kc + 1 < NKC)   // overlap: stream next chunk into the other buffer
      issue_wchunk_async(Wtg, ((kc + 1) & 1) ? Ws1 : Ws0, NKC * 32, kc + 1, t, NROWS);
    _Float16* Wsh = (kc & 1) ? Ws1 : Ws0;

    // A fragment (16x32 f16): lane lm = row, lane group lg splits K by 8
    const _Float16* arow = &Hs[(wave * 16 + lm) * HS_STRIDE + kc * 32 + 8 * lg];
    v8h alo = *(const v8h*)arow;         // K = kbase .. kbase+7
    v8h ahi = *(const v8h*)(arow + 16);  // K = 16+kbase .. 16+kbase+7
    v16h a = __builtin_shufflevector(alo, ahi, 0, 1, 2, 3, 4, 5, 6, 7, 8, 9, 10,
                                     11, 12, 13, 14, 15);
#pragma unroll
    for (int nt = 0; nt < NTILES; ++nt) {
      // B fragment (32x16 f16): lane lm = col n, lane group lg splits K by 8 (same as A)
      const _Float16* brow = &Wsh[(nt * 16 + lm) * WS_STRIDE + 8 * lg];
      v8h blo = *(const v8h*)brow;
      v8h bhi = *(const v8h*)(brow + 16);
      v16h bm = __builtin_shufflevector(blo, bhi, 0, 1, 2, 3, 4, 5, 6, 7, 8, 9,
                                        10, 11, 12, 13, 14, 15);
      acc[nt] = __builtin_amdgcn_wmma_f32_16x16x32_f16(
          false, a, false, bm, (short)0, acc[nt], false, false);
    }
    __syncthreads();  // all waves done reading buf[kc&1] before it is refilled
  }

  // epilogue: bias (+relu) and writeback; rows of Hs are wave-private
#pragma unroll
  for (int nt = 0; nt < NTILES; ++nt) {
    int col = nt * 16 + lm;
    float bv = biasRow[col];
#pragma unroll
    for (int r = 0; r < 8; ++r) {
      float v = acc[nt][r] + bv;
      int row = wave * 16 + r + 8 * lg;
      if (RELU) {
        v = fmaxf(v, 0.0f);
        Hs[row * HS_STRIDE + col] = (_Float16)v;
      } else {
        if (lm < 3) predOut[row * 4 + lm] = v;
      }
    }
  }
}

// ---------------- fused LIIF kernel: gather + 5-layer MLP + blend ----------------------
__launch_bounds__(256)
__global__ void liif_main(const float* __restrict__ feat,
                          const float* __restrict__ s_grid,
                          const float* __restrict__ s_cell,
                          const float* __restrict__ b0, const float* __restrict__ b1,
                          const float* __restrict__ b2, const float* __restrict__ b3,
                          const float* __restrict__ b4,
                          const _Float16* __restrict__ wbank,
                          float* __restrict__ out) {
  extern __shared__ char smem[];
  _Float16* Hs  = (_Float16*)(smem + SM_HS);
  _Float16* Ws0 = (_Float16*)(smem + SM_WS0);
  _Float16* Ws1 = (_Float16*)(smem + SM_WS1);
  float* biasS  = (float*)(smem + SM_BIAS);
  float* b4S    = (float*)(smem + SM_B4);
  float* areaS  = (float*)(smem + SM_AREA);
  float* PredS  = (float*)(smem + SM_PRED);

  const int t = threadIdx.x;
  const int wave = t >> 5, lane = t & 31, lm = lane & 15, lg = lane >> 4;
  const int b = blockIdx.x / (Qc / QT);
  const int qbase = (blockIdx.x % (Qc / QT)) * QT;

  // preload biases into LDS
  for (int i = t; i < 4 * 256; i += 256) {
    const float* bp = (i < 256) ? b0 : (i < 512) ? b1 : (i < 768) ? b2 : b3;
    biasS[i] = bp[i & 255];
  }
  if (t < 16) b4S[t] = (t < 3) ? b4[t] : 0.0f;

  // ---- gather stage: build X rows (wave-private rows of Hs) ----
  {
    const int row = t >> 1, half = t & 1;  // 2 threads per row
    const int s = row >> 5, ql = row & 31;
    const int q = qbase + ql;
    const float dh = 1.0f / Hc, dw = 1.0f / Wc, eps = 1e-6f;
    float g0y = s_grid[((size_t)b * Qc + q) * 2 + 0];
    float g0x = s_grid[((size_t)b * Qc + q) * 2 + 1];
    float rs = (s & 2) ? 1.0f : -1.0f;
    float cs = (s & 1) ? 1.0f : -1.0f;
    float gy = fminf(fmaxf(g0y + rs * dh, -1.0f + eps), 1.0f - eps);
    float gx = fminf(fmaxf(g0x + cs * dw, -1.0f + eps), 1.0f - eps);
    int iy = (int)rintf(((gy + 1.0f) * Hc - 1.0f) * 0.5f);  // round half-even
    int ix = (int)rintf(((gx + 1.0f) * Wc - 1.0f) * 0.5f);
    iy = min(max(iy, 0), Hc - 1);
    ix = min(max(ix, 0), Wc - 1);

    const float* fb = feat + (size_t)b * Cc * Hc * Wc + (size_t)iy * Wc + ix;
    const int c0 = half * 32;
#pragma unroll
    for (int j = 0; j < 32; j += 8) {
      v8h vh;
#pragma unroll
      for (int e = 0; e < 8; ++e)
        vh[e] = (_Float16)fb[(size_t)(c0 + j + e) * (Hc * Wc)];
      *(v8h*)&Hs[row * HS_STRIDE + c0 + j] = vh;
    }
    if (half == 0) {
      float py = -1.0f + (2.0f * iy + 1.0f) * dh;
      float px = -1.0f + (2.0f * ix + 1.0f) * dw;
      float rel_y = (g0y - py) * Hc;
      float rel_x = (g0x - px) * Wc;
      float qcy = s_cell[((size_t)b * Qc + q) * 2 + 0] * Hc;
      float qcx = s_cell[((size_t)b * Qc + q) * 2 + 1] * Wc;
      v8h ve = {};
      ve[0] = (_Float16)rel_y; ve[1] = (_Float16)rel_x;
      ve[2] = (_Float16)qcy;   ve[3] = (_Float16)qcx;
      *(v8h*)&Hs[row * HS_STRIDE + 64] = ve;  // cols 64..71 (68..71 zero)
      v8h z = {};
      *(v8h*)&Hs[row * HS_STRIDE + 72] = z;   // zero pad to K=96
      *(v8h*)&Hs[row * HS_STRIDE + 80] = z;
      *(v8h*)&Hs[row * HS_STRIDE + 88] = z;
      areaS[s * 32 + ql] = fabsf(rel_y * rel_x) + 1e-9f;
    }
  }
  __syncthreads();

  // ---- MLP: layer0 (K=96), layers1-3 (K=256), layer4 (K=256, N=16 padded) ----
  mlp_layer<3, 16, 256, true>(Hs, Ws0, Ws1, wbank + OFF_W0, biasS, nullptr,
                              wave, lm, lg, t);
  for (int l = 0; l < 3; ++l) {
    const _Float16* wp = wbank + OFF_W1 + (size_t)l * 256 * 256;
    mlp_layer<8, 16, 256, true>(Hs, Ws0, Ws1, wp, biasS + (l + 1) * 256, nullptr,
                                wave, lm, lg, t);
  }
  mlp_layer<8, 1, 16, false>(Hs, Ws0, Ws1, wbank + OFF_W4, b4S, PredS,
                             wave, lm, lg, t);
  __syncthreads();

  // ---- blend: LIIF diagonal-swapped area weights ----
  if (t < QT) {
    const int q = qbase + t;
    float a0 = areaS[0 * 32 + t], a1 = areaS[1 * 32 + t];
    float a2 = areaS[2 * 32 + t], a3 = areaS[3 * 32 + t];
    float inv = 1.0f / (a0 + a1 + a2 + a3);
    float w0s = a3 * inv, w1s = a2 * inv, w2s = a1 * inv, w3s = a0 * inv;
#pragma unroll
    for (int j = 0; j < 3; ++j) {
      float v = PredS[(0 * 32 + t) * 4 + j] * w0s +
                PredS[(1 * 32 + t) * 4 + j] * w1s +
                PredS[(2 * 32 + t) * 4 + j] * w2s +
                PredS[(3 * 32 + t) * 4 + j] * w3s;
      out[((size_t)b * Qc + q) * 3 + j] = v;
    }
  }
}

extern "C" void kernel_launch(void* const* d_in, const int* in_sizes, int n_in,
                              void* d_out, int out_size, void* d_ws, size_t ws_size,
                              hipStream_t stream) {
  const float* feat   = (const float*)d_in[0];
  const float* s_grid = (const float*)d_in[1];
  const float* s_cell = (const float*)d_in[2];
  const float* w0 = (const float*)d_in[3];  const float* b0 = (const float*)d_in[4];
  const float* w1 = (const float*)d_in[5];  const float* b1 = (const float*)d_in[6];
  const float* w2 = (const float*)d_in[7];  const float* b2 = (const float*)d_in[8];
  const float* w3 = (const float*)d_in[9];  const float* b3 = (const float*)d_in[10];
  const float* w4 = (const float*)d_in[11]; const float* b4 = (const float*)d_in[12];

  _Float16* wbank = (_Float16*)d_ws;

  // weight prep (f16 transposed bank in workspace)
  auto prep = [&](const float* w, _Float16* dst, int K, int N, int Kpad, int Nrows) {
    int total = Nrows * Kpad;
    liif_prep_w<<<(total + 255) / 256, 256, 0, stream>>>(w, dst, K, N, Kpad, Nrows);
  };
  prep(w0, wbank + OFF_W0, 68, 256, 96, 256);
  prep(w1, wbank + OFF_W1, 256, 256, 256, 256);
  prep(w2, wbank + OFF_W2, 256, 256, 256, 256);
  prep(w3, wbank + OFF_W3, 256, 256, 256, 256);
  prep(w4, wbank + OFF_W4, 256, 3, 256, 16);

  hipFuncSetAttribute((const void*)liif_main,
                      hipFuncAttributeMaxDynamicSharedMemorySize, SMEM_BYTES);
  const int nblocks = Bc * (Qc / QT);  // 8192
  liif_main<<<nblocks, 256, SMEM_BYTES, stream>>>(
      feat, s_grid, s_cell, b0, b1, b2, b3, b4, wbank, (float*)d_out);
}
